// RETAIN_824633721476
// MI455X (gfx1250) — matrix operand order repaired and checked by reading, not compiled, not run on hardware
//
#include <hip/hip_runtime.h>
#include <math.h>

typedef __attribute__((ext_vector_type(16))) _Float16 v16h;
typedef __attribute__((ext_vector_type(8)))  _Float16 v8h;
typedef __attribute__((ext_vector_type(8)))  float    v8f;
typedef __attribute__((ext_vector_type(4)))  int      v4i;

#define B_   64
#define T_   64
#define IND  1024
#define EMB_ 128
#define H_   128
#define D1P  152   // padded temb row stride in f16 elems (304B, 16B-aligned, conflict-free)

// ---- CDNA5 async global->LDS path (probe via __has_builtin; safe fallbacks) ----
#if defined(__has_builtin)
#  if __has_builtin(__builtin_amdgcn_global_load_async_to_lds_b128)
#    define HAVE_ASYNC_B128 1
#  endif
#  if __has_builtin(__builtin_amdgcn_s_wait_asynccnt)
#    define HAVE_WAIT_ASYNC 1
#  endif
#endif

__device__ __forceinline__ void wait_asynccnt0() {
#if defined(HAVE_WAIT_ASYNC)
    __builtin_amdgcn_s_wait_asynccnt(0);
#else
    asm volatile("s_wait_asynccnt 0x0" ::: "memory");
#endif
}

#if defined(HAVE_ASYNC_B128)
typedef __attribute__((address_space(1))) v4i gv4i;   // global int4
typedef __attribute__((address_space(3))) v4i lv4i;   // LDS int4
// 16-byte async copy: global -> LDS, tracked by ASYNCcnt (no VGPR round-trip).
__device__ __forceinline__ void async_copy_b128(const void* g, void* l) {
    __builtin_amdgcn_global_load_async_to_lds_b128(
        (gv4i*)(uintptr_t)g,
        (lv4i*)(uint32_t)(uintptr_t)l,
        0, 0);
}
#endif

__device__ __forceinline__ v8f wmma16(v16h a, v16h b, v8f c) {
    return __builtin_amdgcn_wmma_f32_16x16x32_f16(false, a, false, b, (short)0, c, false, false);
}

// Load a 16x32 f16 A-fragment from LDS (row-major, stride D1P).
// Lane L: M = L%16; elems 0..15 = K = koff + 16*(L/16) .. +15
__device__ __forceinline__ v16h load_afrag(const _Float16* s, int mrow, int half, int koff) {
    const v8h* p0 = (const v8h*)&s[mrow * D1P + koff + half * 16];
    const v8h* p1 = (const v8h*)&s[mrow * D1P + koff + half * 16 + 8];
    v8h lo = *p0, hi = *p1;
    v16h a;
#pragma unroll
    for (int e = 0; e < 8; ++e) { a[e] = lo[e]; a[8 + e] = hi[e]; }
    return a;
}

// ---------------------------------------------------------------------------
// Kernel 1: temb16[b*T+tau][0..127] = f16( x @ W_emb^T ), col 128 = t, 129..151 = 0
// M = 4096 rows, N = 128, K = 1024. One block = 16 rows x 128 cols (8 waves x 1 tile).
// Block 0 also zero-initializes the 304B zero page used by masked async staging.
// ---------------------------------------------------------------------------
__global__ __launch_bounds__(256)
void emb_kernel(const float* __restrict__ x, const float* __restrict__ t,
                const float* __restrict__ W_emb, _Float16* __restrict__ temb16,
                _Float16* __restrict__ zpad)
{
    __shared__ _Float16 sX[16 * 40];
    const int tid  = threadIdx.x;
    const int lane = tid & 31, wv = tid >> 5;
    const int mrow = lane & 15, half = lane >> 4;
    const int row0 = blockIdx.x * 16;

    if (blockIdx.x == 0 && tid < D1P) zpad[tid] = (_Float16)0.f;

    v8f acc = {};
    for (int kc = 0; kc < IND / 32; ++kc) {
        __syncthreads();
        // stage 16x32 chunk of x as f16
        for (int idx = tid; idx < 16 * 32; idx += 256) {
            int m = idx >> 5, c = idx & 31;
            sX[m * 40 + c] = (_Float16)x[(size_t)(row0 + m) * IND + kc * 32 + c];
        }
        __syncthreads();
        // A fragment
        const v8h* p0 = (const v8h*)&sX[mrow * 40 + half * 16];
        const v8h* p1 = (const v8h*)&sX[mrow * 40 + half * 16 + 8];
        v8h lo = *p0, hi = *p1;
        v16h a;
#pragma unroll
        for (int e = 0; e < 8; ++e) { a[e] = lo[e]; a[8 + e] = hi[e]; }
        // B fragment: B[k][n] = W_emb[n][k]; lane holds col n = 16*wv+mrow,
        // 16 consecutive k at kc*32 + 16*half
        v16h b;
        {
            const float* wp = &W_emb[(size_t)(16 * wv + mrow) * IND + kc * 32 + half * 16];
#pragma unroll
            for (int q = 0; q < 16; ++q) b[q] = (_Float16)wp[q];
        }
        acc = wmma16(a, b, acc);
    }
    // store D tile: row m = r + 8*half, col = 16*wv + mrow
#pragma unroll
    for (int r = 0; r < 8; ++r) {
        int m = r + 8 * half;
        temb16[(size_t)(row0 + m) * D1P + 16 * wv + mrow] = (_Float16)acc[r];
    }
    if (tid < 16) {
        int row = row0 + tid;
        temb16[(size_t)row * D1P + 128] = (_Float16)t[row];
        for (int c = 129; c < D1P; ++c) temb16[(size_t)row * D1P + c] = (_Float16)0.f;
    }
}

// ---------------------------------------------------------------------------
// Kernel 2: fused GRU scan + output head.
// Block = 16 sequences (same i, b = b0..b0+15), 8 waves. Runs steps k = 0..i only
// (outputs for k>i are masked to zero by the reference). Weights live in registers
// as f16 WMMA B-fragments; h lives in LDS (f16 mirror + f32 master). Input rows
// are staged with GLOBAL_LOAD_ASYNC_TO_LDS_B128 (masked rows read a zero page).
// ISB==0: GRU A -> alpha softmax head. ISB==1: GRU B -> beta tanh projection head.
// ---------------------------------------------------------------------------
template <int ISB>
__global__ __launch_bounds__(256)
void gru_kernel(const _Float16* __restrict__ temb16,
                const _Float16* __restrict__ zpad,
                const float* __restrict__ t, const int* __restrict__ lengths,
                const float* __restrict__ Wih, const float* __restrict__ Whh,
                const float* __restrict__ bih, const float* __restrict__ bhh,
                const float* __restrict__ Wafc, const float* __restrict__ bafc,
                const float* __restrict__ Wbfc, const float* __restrict__ bbfc,
                float* __restrict__ out_alpha, float* __restrict__ out_beta)
{
    __shared__ _Float16 sIn[16 * D1P];   // current-step input rows (f16)
    __shared__ _Float16 sH16[16 * D1P];  // h state, f16 mirror (A operand)
    __shared__ float    sRZ[16 * 260];   // r,z gate pre-activations (x+h+biases)
    __shared__ float    sNX[16 * 132];   // n-gate input part (+bih)
    __shared__ float    sNH[16 * 132];   // n-gate recurrent part (+bhh)
    __shared__ float    sHF[16 * 128];   // h state, f32 master
    __shared__ float    sT[16];          // t-column values (f32)
    __shared__ int      sLenV[16];
    __shared__ float    sWafc[128];
    __shared__ float    sPart[16 * 16];
    __shared__ float    sPreA[16 * 64];

    const int tid  = threadIdx.x;
    const int lane = tid & 31, wv = tid >> 5;
    const int mrow = lane & 15, half = lane >> 4;
    const int s0    = blockIdx.x * 16;   // flat sequence index base (= i*B + b0)
    const int i_idx = s0 / B_;
    const int b0    = s0 % B_;

    // ---- load persistent weight fragments into registers (once) ----
    v16h fih[3][4], fhh[3][4];
    float wt[3], bih_v[3], bhh_v[3];
#pragma unroll
    for (int jt = 0; jt < 3; ++jt) {
        int n = (3 * wv + jt) * 16 + mrow;
#pragma unroll
        for (int c = 0; c < 4; ++c) {
            const float* p = &Wih[(size_t)n * 129 + c * 32 + half * 16];
            const float* q = &Whh[(size_t)n * 128 + c * 32 + half * 16];
#pragma unroll
            for (int e = 0; e < 16; ++e) {
                fih[jt][c][e] = (_Float16)p[e];
                fhh[jt][c][e] = (_Float16)q[e];
            }
        }
        wt[jt]    = Wih[(size_t)n * 129 + 128];
        bih_v[jt] = bih[n];
        bhh_v[jt] = bhh[n];
    }
    v16h ffc[4];
    float bfc_v = 0.f, bafc0 = 0.f;
    if (ISB) {
        int e = 16 * wv + mrow;
#pragma unroll
        for (int c = 0; c < 4; ++c) {
            const float* p = &Wbfc[(size_t)e * 128 + c * 32 + half * 16];
#pragma unroll
            for (int q = 0; q < 16; ++q) ffc[c][q] = (_Float16)p[q];
        }
        bfc_v = bbfc[e];
    } else {
        for (int j = tid; j < 128; j += 256) sWafc[j] = Wafc[j];
        bafc0 = bafc[0];
    }

    // ---- init state ----
    for (int idx = tid; idx < 16 * 128; idx += 256) sHF[idx] = 0.f;
    for (int idx = tid; idx < 16 * D1P; idx += 256) sH16[idx] = (_Float16)0.f;
    if (tid < 16) sLenV[tid] = lengths[b0 + tid];

    const int kend = i_idx;  // only steps 0..i contribute to masked outputs
    for (int k = 0; k <= kend; ++k) {
        const int tau = i_idx - k;
        __syncthreads();
        // ---- stage input rows: temb16[b0+m][tau][:] (masked rows -> zero page) ----
#if defined(HAVE_ASYNC_B128)
        for (int idx = tid; idx < 16 * 19; idx += 256) {   // 19 x 16B chunks per row
            int m = idx / 19, c = idx % 19;
            bool live = (i_idx < sLenV[m]);
            const _Float16* src = live
                ? &temb16[((size_t)(b0 + m) * T_ + tau) * D1P + c * 8]
                : &zpad[c * 8];
            if (live && tau > 0 && c == 0)
                __builtin_prefetch(src - D1P, 0, 1);       // warm L2 for next step
            async_copy_b128(src, &sIn[m * D1P + c * 8]);
        }
        wait_asynccnt0();
#else
        for (int idx = tid; idx < 16 * (D1P / 2); idx += 256) {
            int m = idx / (D1P / 2), c = idx % (D1P / 2);
            const uint32_t* src =
                (const uint32_t*)&temb16[((size_t)(b0 + m) * T_ + tau) * D1P];
            ((uint32_t*)sIn)[m * (D1P / 2) + c] = (i_idx < sLenV[m]) ? src[c] : 0u;
        }
#endif
        if (tid < 16)
            sT[tid] = (i_idx < sLenV[tid]) ? t[(size_t)(b0 + tid) * T_ + tau] : 0.f;
        __syncthreads();

        // ---- A fragments for input and h ----
        v16h aI[4], aH[4];
#pragma unroll
        for (int c = 0; c < 4; ++c) {
            aI[c] = load_afrag(sIn,  mrow, half, c * 32);
            aH[c] = load_afrag(sH16, mrow, half, c * 32);
        }
        float tvals[8];
#pragma unroll
        for (int r = 0; r < 8; ++r) tvals[r] = sT[r + 8 * half];

        // ---- gate GEMMs: 3 N-tiles per wave, 24 WMMAs per wave per step ----
#pragma unroll
        for (int jt = 0; jt < 3; ++jt) {
            v8f xacc = {}, hacc = {};
#pragma unroll
            for (int c = 0; c < 4; ++c) {
                xacc = wmma16(aI[c], fih[jt][c], xacc);
                hacc = wmma16(aH[c], fhh[jt][c], hacc);
            }
            int n0   = (3 * wv + jt) * 16;
            int ncol = n0 + mrow;
            if (n0 < 256) {  // r,z gates: combine x+h parts
#pragma unroll
                for (int r = 0; r < 8; ++r) {
                    int m = r + 8 * half;
                    sRZ[m * 260 + ncol] =
                        xacc[r] + tvals[r] * wt[jt] + bih_v[jt] + hacc[r] + bhh_v[jt];
                }
            } else {         // n gate: keep parts separate (r multiplies only recurrent)
                int nc = ncol - 256;
#pragma unroll
                for (int r = 0; r < 8; ++r) {
                    int m = r + 8 * half;
                    sNX[m * 132 + nc] = xacc[r] + tvals[r] * wt[jt] + bih_v[jt];
                    sNH[m * 132 + nc] = hacc[r] + bhh_v[jt];
                }
            }
        }
        __syncthreads();

        // ---- elementwise GRU update: each thread owns (m, 8 consecutive j) ----
        {
            int m  = tid >> 4;
            int jb = (tid & 15) * 8;
            float partial = 0.f;
#pragma unroll
            for (int q = 0; q < 8; ++q) {
                int j = jb + q;
                float rr   = 1.f / (1.f + __expf(-sRZ[m * 260 + j]));
                float zz   = 1.f / (1.f + __expf(-sRZ[m * 260 + 128 + j]));
                float nn   = tanhf(sNX[m * 132 + j] + rr * sNH[m * 132 + j]);
                float hold = sHF[m * 128 + j];
                float hnew = (1.f - zz) * nn + zz * hold;
                sHF[m * 128 + j]  = hnew;
                sH16[m * D1P + j] = (_Float16)hnew;
                if (!ISB) partial += hnew * sWafc[j];
            }
            if (!ISB) sPart[m * 16 + (tid & 15)] = partial;
        }
        __syncthreads();

        if (!ISB) {
            // alpha head: reduce dot(h, W_afc) per row, bank pre-softmax logits
            if (tid < 16) {
                float s = 0.f;
#pragma unroll
                for (int q = 0; q < 16; ++q) s += sPart[tid * 16 + q];
                sPreA[tid * 64 + k] = s + bafc0;
            }
        } else {
            // beta head: (16x128 h) x W_bfc^T, one 16-col tile per wave
            v16h aH2[4];
#pragma unroll
            for (int c = 0; c < 4; ++c) aH2[c] = load_afrag(sH16, mrow, half, c * 32);
            v8f facc = {};
#pragma unroll
            for (int c = 0; c < 4; ++c) facc = wmma16(aH2[c], ffc[c], facc);
            int e = 16 * wv + mrow;
#pragma unroll
            for (int r = 0; r < 8; ++r) {
                int m = r + 8 * half;
                size_t off = (((size_t)(s0 + m)) * T_ + k) * EMB_ + e;
                out_beta[off] = tanhf(facc[r] + bfc_v);
            }
        }
    }
    __syncthreads();

    if (!ISB) {
        // masked softmax over k = 0..i, zeros beyond
        if (tid < 16) {
            int m = tid;
            float mx = -1e30f;
            for (int k = 0; k <= kend; ++k) mx = fmaxf(mx, sPreA[m * 64 + k]);
            float sum = 0.f;
            for (int k = 0; k <= kend; ++k) sum += __expf(sPreA[m * 64 + k] - mx);
            float inv = 1.f / sum;
            size_t base = ((size_t)(s0 + m)) * T_;
            for (int k = 0; k <= kend; ++k)
                out_alpha[base + k] = __expf(sPreA[m * 64 + k] - mx) * inv;
            for (int k = kend + 1; k < T_; ++k) out_alpha[base + k] = 0.f;
        }
    } else {
        // zero-fill masked beta region k > i
        int nfill = (T_ - 1 - kend) * 16 * EMB_;
        for (int idx = tid; idx < nfill; idx += 256) {
            int kk  = kend + 1 + idx / (16 * EMB_);
            int rem = idx % (16 * EMB_);
            int m = rem / EMB_, e = rem % EMB_;
            out_beta[(((size_t)(s0 + m)) * T_ + kk) * EMB_ + e] = 0.f;
        }
    }
}

extern "C" void kernel_launch(void* const* d_in, const int* in_sizes, int n_in,
                              void* d_out, int out_size, void* d_ws, size_t ws_size,
                              hipStream_t stream) {
    const float* x       = (const float*)d_in[0];
    const float* t       = (const float*)d_in[1];
    const int*   lengths = (const int*)d_in[2];
    const float* W_emb   = (const float*)d_in[3];
    const float* Wih_a   = (const float*)d_in[4];
    const float* Whh_a   = (const float*)d_in[5];
    const float* bih_a   = (const float*)d_in[6];
    const float* bhh_a   = (const float*)d_in[7];
    const float* Wih_b   = (const float*)d_in[8];
    const float* Whh_b   = (const float*)d_in[9];
    const float* bih_b   = (const float*)d_in[10];
    const float* bhh_b   = (const float*)d_in[11];
    const float* W_afc   = (const float*)d_in[12];
    const float* b_afc   = (const float*)d_in[13];
    const float* W_bfc   = (const float*)d_in[14];
    const float* b_bfc   = (const float*)d_in[15];

    float* out_alpha = (float*)d_out;                       // (T,B,T,1)
    float* out_beta  = out_alpha + (size_t)T_ * B_ * T_;    // (T,B,T,EMB)
    _Float16* temb16 = (_Float16*)d_ws;                     // (B*T, D1P) f16
    _Float16* zpad   = temb16 + (size_t)T_ * B_ * D1P;      // 304B zero page

    emb_kernel<<<dim3(256), dim3(256), 0, stream>>>(x, t, W_emb, temb16, zpad);
    gru_kernel<0><<<dim3(256), dim3(256), 0, stream>>>(
        temb16, zpad, t, lengths, Wih_a, Whh_a, bih_a, bhh_a,
        W_afc, b_afc, W_bfc, b_bfc, out_alpha, out_beta);
    gru_kernel<1><<<dim3(256), dim3(256), 0, stream>>>(
        temb16, zpad, t, lengths, Wih_b, Whh_b, bih_b, bhh_b,
        W_afc, b_afc, W_bfc, b_bfc, out_alpha, out_beta);
}
